// VolumetricRenderer_89446988907044
// MI455X (gfx1250) — matrix operand order repaired and checked by reading, not compiled, not run on hardware
//
#include <hip/hip_runtime.h>

// ---------------------------------------------------------------------------
// Triplane NeRF renderer for MI455X (gfx1250, wave32, WMMA).
//  - planes pre-transposed to [plane][y][x][feat] f16 (L2-resident, 3MB)
//  - W1^T / W2^T pre-packed into a contiguous-K (k=16h+s) 16x16x32 f16 A-operand lane layout
//  - layer1 D (f32) -> layer2 B (f16) via cvt_pkrtz + packed f16 ReLU + 1 shfl
//  - per-ray volume integration lives entirely in lane == ray
// ---------------------------------------------------------------------------

typedef _Float16 half_t;
typedef __attribute__((ext_vector_type(2)))  _Float16 h2;
typedef __attribute__((ext_vector_type(16))) _Float16 v16h;
typedef __attribute__((ext_vector_type(8)))  float    v8f;

#define RESO  128
#define DF    32
#define HH    128
#define WW    128
#define SS    64
#define NCAM  4
#define NEARP 0.5f
#define FARP  3.5f

static __device__ __forceinline__ v16h splat16(_Float16 v) {
  v16h r;
  #pragma unroll
  for (int i = 0; i < 16; ++i) r[i] = v;
  return r;
}

static __device__ __forceinline__ float softplusf(float x) {
  // max(x,0) via single v_med3; log1p(exp(-|x|)) via fast log/exp
  float mx = __builtin_amdgcn_fmed3f(x, 0.f, 3.0e38f);
  return mx + __logf(1.f + __expf(-fabsf(x)));
}
static __device__ __forceinline__ float sigmoidf(float x) {
  return 1.f / (1.f + __expf(-x));
}

// f32 pair -> f16 pair (v_cvt_pk_rtz) then packed ReLU (v_pk_max_num_f16 vs 0).
static __device__ __forceinline__ unsigned relu_pack(float a, float b) {
  unsigned u = __builtin_bit_cast(unsigned, __builtin_amdgcn_cvt_pkrtz(a, b));
  unsigned r;
  asm("v_pk_max_num_f16 %0, %1, 0" : "=v"(r) : "v"(u));
  return r;
}

// One bilinear triplane tap: 4 contiguous 32B corner loads, 2-step pk-f16 lerp.
// Corner addresses derived incrementally from o00 (32-bit offsets off SGPR base).
static __device__ __forceinline__ v16h sample_plane(const half_t* __restrict__ pl,
                                                    float a, float b) {
  float gx = fminf(fmaxf((a + 1.f) * 0.5f * (float)(RESO - 1), 0.f), (float)(RESO - 1));
  float gy = fminf(fmaxf((b + 1.f) * 0.5f * (float)(RESO - 1), 0.f), (float)(RESO - 1));
  float fx0 = floorf(gx), fy0 = floorf(gy);
  int x0 = (int)fx0, y0 = (int)fy0;
  int x1 = min(x0 + 1, RESO - 1), y1 = min(y0 + 1, RESO - 1);
  _Float16 wx = (_Float16)(gx - fx0);
  _Float16 wy = (_Float16)(gy - fy0);
  int o00 = ((y0 << 7) + x0) << 5;    // element offset of 00 corner
  int dxo = (x1 - x0) << 5;           // 0 or 32 elements
  int dyo = (y1 - y0) << 12;          // 0 or 4096 elements
  v16h f00 = *(const v16h*)(pl + o00);
  v16h f01 = *(const v16h*)(pl + (o00 + dxo));
  v16h f10 = *(const v16h*)(pl + (o00 + dyo));
  v16h f11 = *(const v16h*)(pl + (o00 + dyo + dxo));
  v16h vwx = splat16(wx), vwy = splat16(wy);
  v16h top = f00 + (f01 - f00) * vwx;
  v16h bot = f10 + (f11 - f10) * vwx;
  return top + (bot - top) * vwy;
}

// planes (3,32,128,128) f32  ->  planesT [3][y][x][f] f16 (feature-contiguous)
__global__ __launch_bounds__(256) void vr_prep_planes(const float* __restrict__ planes,
                                                      half_t* __restrict__ planesT) {
  int idx = blockIdx.x * blockDim.x + threadIdx.x;
  if (idx >= 3 * RESO * RESO) return;
  int x = idx & (RESO - 1);
  int y = (idx >> 7) & (RESO - 1);
  int p = idx >> 14;
  half_t* dst = planesT + (idx << 5);
  const float* src = planes + ((p * DF) << 14) + (y << 7) + x;
  #pragma unroll
  for (int f = 0; f < DF; ++f) dst[f] = (half_t)src[f << 14];
}

// Pack W1^T (8 tiles, M=hidden) and W2^T (4 K-tiles, M=chan, zero-padded rows)
// into a contiguous-K 16x16x32 f16 A-operand lane layout (B uses the same k order):
//   lane l<16 : row M=l,  halfs 0..15 -> K=0..15
//   lane 16+l : row M=l,  halfs 0..15 -> K=16..31
__global__ __launch_bounds__(256) void vr_prep_weights(const float* __restrict__ W1,
                                                       const float* __restrict__ W2,
                                                       half_t* __restrict__ w1a,
                                                       half_t* __restrict__ w2a) {
  int idx = blockIdx.x * blockDim.x + threadIdx.x;
  if (idx >= 8 * 32 * 16 + 4 * 32 * 16) return;
  if (idx < 8 * 32 * 16) {
    int g = idx >> 9;
    int lane = (idx >> 4) & 31;
    int i = idx & 15;
    int l = lane & 15, hi = lane >> 4;
    int kk = i + (hi ? 16 : 0);
    w1a[idx] = (half_t)W1[kk * 128 + 16 * g + l];   // W1 is (32,128): W1[k][hidden]
  } else {
    int j = idx - 8 * 32 * 16;
    int t = j >> 9;
    int lane = (j >> 4) & 31;
    int i = j & 15;
    int l = lane & 15, hi = lane >> 4;
    int kk = i + (hi ? 16 : 0);
    int K = 32 * t + kk;
    w2a[j] = (l < 4) ? (half_t)W2[K * 4 + l] : (half_t)0.f;  // W2 is (128,4)
  }
}

__global__ __launch_bounds__(256) void vr_render(const float* __restrict__ intr,
                                                 const float* __restrict__ extr,
                                                 const half_t* __restrict__ planesT,
                                                 const half_t* __restrict__ w1a,
                                                 const half_t* __restrict__ w2a,
                                                 const float* __restrict__ b1,
                                                 const float* __restrict__ b2,
                                                 float* __restrict__ out) {
  const int tid  = blockIdx.x * blockDim.x + threadIdx.x;
  const int lane = threadIdx.x & 31;
  const int l    = lane & 15;
  const int hi   = lane >> 4;           // 0: feats 0..15, 1: feats 16..31
  const int wave = tid >> 5;
  const int ray  = (wave << 4) + l;     // 16 rays per wave; lane pair (l, l+16)
  const int cam  = ray >> 14;
  const int pix  = ray & (HH * WW - 1);
  const float vy = (float)(pix >> 7);
  const float ux = (float)(pix & (WW - 1));

  // --- camera / ray setup (duplicated across the lane pair; trivial cost) ---
  const float* In = intr + cam * 9;
  float fx = In[0] * 0.5f, cx = In[2] * 0.5f;
  float fy = In[4] * 0.5f, cy = In[5] * 0.5f;
  float dcx = (ux - cx) / fx;
  float dcy = (vy - cy) / fy;
  const float* E = extr + cam * 16;     // row-major 4x4; R=E[:3,:3], t=E[:3,3]
  float ox = -(E[0] * E[3] + E[4] * E[7] + E[8]  * E[11]);
  float oy = -(E[1] * E[3] + E[5] * E[7] + E[9]  * E[11]);
  float oz = -(E[2] * E[3] + E[6] * E[7] + E[10] * E[11]);
  float ddx = E[0] * dcx + E[4] * dcy + E[8];
  float ddy = E[1] * dcx + E[5] * dcy + E[9];
  float ddz = E[2] * dcx + E[6] * dcy + E[10];
  float inv = rsqrtf(ddx * ddx + ddy * ddy + ddz * ddz);
  ddx *= inv; ddy *= inv; ddz *= inv;

  // --- hoisted WMMA operands (A tiles of W1^T / W2^T, bias as C-init) ---
  v16h a1[8];
  #pragma unroll
  for (int g = 0; g < 8; ++g) a1[g] = *(const v16h*)(w1a + ((g * 32 + lane) << 4));
  v16h a2[4];
  #pragma unroll
  for (int t = 0; t < 4; ++t) a2[t] = *(const v16h*)(w2a + ((t * 32 + lane) << 4));
  v8f b1v[8];
  #pragma unroll
  for (int g = 0; g < 8; ++g) b1v[g] = *(const v8f*)(b1 + 16 * g + hi * 8);
  const float b2v0 = b2[0], b2v1 = b2[1], b2v2 = b2[2], b2v3 = b2[3];

  const int fbase = hi << 4;
  const half_t* p0 = planesT + fbase;
  const half_t* p1 = planesT + RESO * RESO * DF + fbase;
  const half_t* p2 = planesT + 2 * RESO * RESO * DF + fbase;

  const float dT = (FARP - NEARP) / (float)(SS - 1);
  float T = 1.f, aR = 0.f, aG = 0.f, aB = 0.f;

  for (int s = 0; s < SS; ++s) {
    float tt = NEARP + dT * (float)s;
    float px = ox + tt * ddx;
    float py = oy + tt * ddy;
    float pz = oz + tt * ddz;

    // feats^T B-operand: lane pair splits the 32 features, no LDS needed
    v16h feats = sample_plane(p0, px, py)
               + sample_plane(p1, px, pz)
               + sample_plane(p2, py, pz);

    // ---- layer 1: h^T(16 hidden x 16 rays) per group, bias in C-init ----
    v8f acc[8];
    #pragma unroll
    for (int g = 0; g < 8; ++g) acc[g] = b1v[g];
    #pragma unroll
    for (int g = 0; g < 8; ++g)
      acc[g] = __builtin_amdgcn_wmma_f32_16x16x32_f16(false, a1[g], false, feats,
                                                      (short)0, acc[g], false, false);

    // ---- f32->f16 pair pack + packed ReLU ----
    unsigned pku[8][4];
    #pragma unroll
    for (int g = 0; g < 8; ++g) {
      #pragma unroll
      for (int i = 0; i < 4; ++i)
        pku[g][i] = relu_pack(acc[g][2 * i + 0], acc[g][2 * i + 1]);
    }

    // ---- layer 2: out^T = W2^T x h^T; B via 1 shfl + 3 selects per pair ----
    v8f d2;
    #pragma unroll
    for (int i = 0; i < 8; ++i) d2[i] = 0.f;
    d2[0] = b2v0; d2[1] = b2v1; d2[2] = b2v2; d2[3] = b2v3;
    #pragma unroll
    for (int t = 0; t < 4; ++t) {
      unsigned bw[8];
      #pragma unroll
      for (int i = 0; i < 4; ++i) {
        unsigned A  = pku[2 * t][i];        // h rows 32t+{0..7 | 8..15} (by half-wave)
        unsigned Bv = pku[2 * t + 1][i];    // h rows 32t+16+{0..7 | 8..15}
        unsigned Y  = hi ? A : Bv;          // value my partner needs
        unsigned sw = (unsigned)__shfl_xor((int)Y, 16, 32);
        bw[i]     = hi ? sw : A;            // K = 32t + 2i, 2i+1   (per-lane view)
        bw[4 + i] = hi ? Bv : sw;           // K = 32t + 8+2i, ...
      }
      v16h bt;
      #pragma unroll
      for (int i = 0; i < 8; ++i) {
        h2 hh = __builtin_bit_cast(h2, bw[i]);
        bt[2 * i]     = hh[0];
        bt[2 * i + 1] = hh[1];
      }
      d2 = __builtin_amdgcn_wmma_f32_16x16x32_f16(false, a2[t], false, bt,
                                                  (short)0, d2, false, false);
    }

    // ---- volume integration: lane r owns ray r (lanes 16..31 compute junk) ----
    float sigma = softplusf(d2[3]);
    float delta = (s == SS - 1) ? 1e10f : dT;
    float e = __expf(-sigma * delta);
    float w = T * (1.f - e);
    aR += w * sigmoidf(d2[0]);
    aG += w * sigmoidf(d2[1]);
    aB += w * sigmoidf(d2[2]);
    T *= e;
  }

  if (hi == 0) {
    int base = ((cam * 3) << 14) + pix;     // out (1,4,3,128,128)
    out[base]             = aR;
    out[base + (1 << 14)] = aG;
    out[base + (2 << 14)] = aB;
  }
}

extern "C" void kernel_launch(void* const* d_in, const int* in_sizes, int n_in,
                              void* d_out, int out_size, void* d_ws, size_t ws_size,
                              hipStream_t stream) {
  const float* intr   = (const float*)d_in[0];
  const float* extr   = (const float*)d_in[1];
  const float* planes = (const float*)d_in[2];
  const float* W1     = (const float*)d_in[3];
  const float* b1     = (const float*)d_in[4];
  const float* W2     = (const float*)d_in[5];
  const float* b2     = (const float*)d_in[6];

  half_t* planesT = (half_t*)d_ws;                         // 3*128*128*32 f16 = 3MB
  half_t* w1a = planesT + 3 * RESO * RESO * DF;            // 8*32*16 f16
  half_t* w2a = w1a + 8 * 32 * 16;                         // 4*32*16 f16

  vr_prep_planes<<<(3 * RESO * RESO + 255) / 256, 256, 0, stream>>>(planes, planesT);
  vr_prep_weights<<<(8 * 32 * 16 + 4 * 32 * 16 + 255) / 256, 256, 0, stream>>>(W1, W2, w1a, w2a);

  const int waves  = (NCAM * HH * WW) / 16;                // 4096 waves, 16 rays each
  const int blocks = waves * 32 / 256;                     // 512 blocks of 256
  vr_render<<<blocks, 256, 0, stream>>>(intr, extr, planesT, w1a, w2a, b1, b2,
                                        (float*)d_out);
}
